// ResidualAttentionBlock_46755013984856
// MI455X (gfx1250) — matrix operand order repaired and checked by reading, not compiled, not run on hardware
//
#include <hip/hip_runtime.h>
#include <stdint.h>

// ---------------- problem constants ----------------
#define DM    1024
#define NHEAD 16
#define HD    64
#define TT    8
#define SS    14
#define LTOK  196          // S*S
#define LT    1568         // L*T
#define NB    8
#define MQ_T  12544        // LT*NB   (temporal token rows)
#define MQ_S  12608        // 197*64  (spatial token rows)
#define MALL  12552        // (1+LT)*NB

typedef __bf16 v16bf __attribute__((ext_vector_type(16)));
typedef float  v8f   __attribute__((ext_vector_type(8)));

union FragU { uint32_t u[8]; v16bf v; };
union AccU  { float f[8];    v8f  v; };

__device__ __forceinline__ uint16_t f2bf(float f) {
    uint32_t u = __float_as_uint(f);
    u = (u + 0x7FFFu + ((u >> 16) & 1u)) >> 16;
    return (uint16_t)u;
}
__device__ __forceinline__ float bf2f(uint16_t h) {
    return __uint_as_float(((uint32_t)h) << 16);
}
__device__ __forceinline__ int kpA(int v, int hi) { return ((v >> 2) << 3) + (v & 3) + (hi << 2); }
__device__ __forceinline__ int kpB(int v, int hi) { return ((v >> 2) << 3) + (v & 3) + (hi << 2); }

// ---- CDNA5 async global->LDS (ASYNCcnt) helpers ----
// 16B per lane, LDS dest = low 32 bits of generic shared pointer (wave-relative).
__device__ __forceinline__ void async_ld16(uint32_t lds_off, const void* gaddr) {
    asm volatile("global_load_async_to_lds_b128 %0, %1, off"
                 :: "v"(lds_off), "v"((uint64_t)(uintptr_t)gaddr)
                 : "memory");
}
__device__ __forceinline__ void wait_async0() {
    asm volatile("s_wait_asynccnt 0x0" ::: "memory");
}

// ---------------- generic helpers ----------------
__global__ __launch_bounds__(256) void cast_f32_bf16(const float* __restrict__ in,
                                                     uint16_t* __restrict__ out, int n) {
    int i = blockIdx.x * 256 + threadIdx.x;
    if (i < n) out[i] = f2bf(in[i]);
}

__global__ __launch_bounds__(256) void copy_f32(const float* __restrict__ in,
                                                float* __restrict__ out, int n) {
    int i = blockIdx.x * 256 + threadIdx.x;
    if (i < n) out[i] = in[i];
}

// depthwise 3x3x3 conv positional embedding:  Xc[1+lt,n,c] = v + conv(v) + pb
__global__ __launch_bounds__(256) void conv_pos(const float* __restrict__ xin,
                                                const float* __restrict__ pw,
                                                const float* __restrict__ pb,
                                                float* __restrict__ Xc) {
    size_t idx = (size_t)blockIdx.x * 256 + threadIdx.x;
    if (idx >= (size_t)LT * NB * DM) return;
    int c  = (int)(idx & 1023);
    int n  = (int)((idx >> 10) & 7);
    int lt = (int)(idx >> 13);
    int t  = lt & 7;         // lt = l*T + t
    int l  = lt >> 3;
    int hh = l / SS, ww = l % SS;
    const float* wc = pw + c * 27;
    float acc = 0.f;
    #pragma unroll
    for (int dt = 0; dt < 3; dt++) {
        int tt2 = t + dt - 1; if (tt2 < 0 || tt2 >= TT) continue;
        #pragma unroll
        for (int dh = 0; dh < 3; dh++) {
            int h2 = hh + dh - 1; if (h2 < 0 || h2 >= SS) continue;
            #pragma unroll
            for (int dw = 0; dw < 3; dw++) {
                int w2 = ww + dw - 1; if (w2 < 0 || w2 >= SS) continue;
                int lt2 = (h2 * SS + w2) * TT + tt2;
                acc += wc[dt * 9 + dh * 3 + dw] *
                       xin[(size_t)((1 + lt2) * NB + n) * DM + c];
            }
        }
    }
    size_t o = (size_t)((1 + lt) * NB + n) * DM + c;
    Xc[o] = xin[o] + acc + pb[c];
}

// fp32 LayerNorm over 1024 cols -> bf16 row
__device__ __forceinline__ void ln_body(const float* __restrict__ src,
                                        const float* __restrict__ g,
                                        const float* __restrict__ bt,
                                        uint16_t* __restrict__ dst) {
    int tid = threadIdx.x;
    float v[4]; float s = 0.f, s2 = 0.f;
    #pragma unroll
    for (int i = 0; i < 4; i++) {
        float x = src[tid + i * 256];
        v[i] = x; s += x; s2 += x * x;
    }
    #pragma unroll
    for (int m = 16; m >= 1; m >>= 1) { s += __shfl_xor(s, m, 32); s2 += __shfl_xor(s2, m, 32); }
    __shared__ float red[16];
    int wid = tid >> 5, lane = tid & 31;
    if (lane == 0) { red[wid] = s; red[8 + wid] = s2; }
    __syncthreads();
    s = 0.f; s2 = 0.f;
    #pragma unroll
    for (int i = 0; i < 8; i++) { s += red[i]; s2 += red[8 + i]; }
    float mean = s * (1.f / DM);
    float var  = s2 * (1.f / DM) - mean * mean;
    float inv  = rsqrtf(var + 1e-5f);
    #pragma unroll
    for (int i = 0; i < 4; i++) {
        int c = tid + i * 256;
        dst[c] = f2bf((v[i] - mean) * inv * g[c] + bt[c]);
    }
}

__global__ __launch_bounds__(256) void ln_rows(const float* __restrict__ in,
                                               const float* __restrict__ g,
                                               const float* __restrict__ bt,
                                               uint16_t* __restrict__ out, int R) {
    int r = blockIdx.x;
    if (r >= R) return;
    ln_body(in + (size_t)r * DM, g, bt, out + (size_t)r * DM);
}

// build xs rows (b=t*8+n major, s in [0,197)) with LN1 -> bf16
__global__ __launch_bounds__(256) void pack_xs_ln(const float* __restrict__ Xc,
                                                  const float* __restrict__ XTF,
                                                  const float* __restrict__ g,
                                                  const float* __restrict__ bt,
                                                  uint16_t* __restrict__ out) {
    int rb = blockIdx.x;                  // 0..12607
    int b = rb / 197, s = rb % 197;
    int t = b >> 3, n = b & 7;
    const float* src;
    if (s == 0) src = Xc + (size_t)n * DM;                              // cls token
    else        src = XTF + (size_t)(((s - 1) * TT + t) * NB + n) * DM; // xt_full row
    ln_body(src, g, bt, out + (size_t)rb * DM);
}

// ---------------- WMMA NT GEMM: C[M,N] = A[M,K](bf16) * W[N,K]^T(bf16) + bias ----------------
// Tiles staged with GLOBAL_LOAD_ASYNC_TO_LDS_B128 (ASYNCcnt), double buffered.
// MODE 0: f32 out    1: f32 out + residual    2: bf16 quickgelu out    3: bf16 out
template <int MODE>
__global__ __launch_bounds__(256) void gemm_nt(const uint16_t* __restrict__ A,
                                               const uint16_t* __restrict__ W,
                                               const float* __restrict__ bias,
                                               const float* __restrict__ resid,
                                               float* __restrict__ outF,
                                               uint16_t* __restrict__ outB,
                                               int M, int N, int K) {
    __shared__ uint32_t As[2][128 * 20];
    __shared__ uint32_t Bs[2][128 * 20];
    const int tid = threadIdx.x;
    const int wid = tid >> 5, lane = tid & 31;
    const int l15 = lane & 15, hi = lane >> 4;
    const int wm = wid >> 2, wn = wid & 3;
    const int bm = blockIdx.y, bn = blockIdx.x;
    const int half = tid & 1;
    const int rowA = bm * 128 + (tid >> 1);
    const int rowW = bn * 128 + (tid >> 1);
    const uint32_t* Ag = (const uint32_t*)A;
    const uint32_t* Wg = (const uint32_t*)W;
    const int K2 = K >> 1;
    const int lo = (tid >> 1) * 20 + half * 8;   // 16B-aligned LDS slot (80B row stride)

    float acc[8][8];
    #pragma unroll
    for (int i = 0; i < 8; i++)
        #pragma unroll
        for (int j = 0; j < 8; j++) acc[i][j] = 0.f;

    // zero LDS rows for out-of-range M (async loads are never issued for them)
    if (rowA >= M) {
        #pragma unroll
        for (int j = 0; j < 8; j++) { As[0][lo + j] = 0u; As[1][lo + j] = 0u; }
    }

    auto issue = [&](int kt, int buf) {
        int kc = kt * 16 + half * 8;
        if (rowA < M) {
            const uint32_t* ga = Ag + (size_t)rowA * K2 + kc;
            uint32_t la = (uint32_t)(uintptr_t)&As[buf][lo];
            async_ld16(la, ga);
            async_ld16(la + 16, ga + 4);
        }
        const uint32_t* gw = Wg + (size_t)rowW * K2 + kc;
        uint32_t lb = (uint32_t)(uintptr_t)&Bs[buf][lo];
        async_ld16(lb, gw);
        async_ld16(lb + 16, gw + 4);
    };

    auto mma = [&](int buf) {
        FragU a[4], b[2];
        #pragma unroll
        for (int tm = 0; tm < 4; tm++) {
            int r = wm * 64 + tm * 16 + l15;
            #pragma unroll
            for (int v = 0; v < 8; v++) a[tm].u[v] = As[buf][r * 20 + kpA(v, hi)];
        }
        #pragma unroll
        for (int tn = 0; tn < 2; tn++) {
            int r = wn * 32 + tn * 16 + l15;
            #pragma unroll
            for (int v = 0; v < 8; v++) b[tn].u[v] = Bs[buf][r * 20 + kpB(v, hi)];
        }
        #pragma unroll
        for (int tm = 0; tm < 4; tm++)
            #pragma unroll
            for (int tn = 0; tn < 2; tn++) {
                AccU c;
                #pragma unroll
                for (int r = 0; r < 8; r++) c.f[r] = acc[tm * 2 + tn][r];
                c.v = __builtin_amdgcn_wmma_f32_16x16x32_bf16(
                          false, a[tm].v, false, b[tn].v, (short)0, c.v, false, false);
                #pragma unroll
                for (int r = 0; r < 8; r++) acc[tm * 2 + tn][r] = c.f[r];
            }
    };

    const int nk = K >> 5;
    issue(0, 0);
    for (int kt = 0; kt < nk; kt++) {
        wait_async0();               // own async writes into buf (kt&1) done
        __syncthreads();             // everyone's writes visible; prior reads done
        if (kt + 1 < nk) issue(kt + 1, (kt + 1) & 1);  // overlap with compute
        mma(kt & 1);
    }

    #pragma unroll
    for (int tm = 0; tm < 4; tm++)
        #pragma unroll
        for (int tn = 0; tn < 2; tn++) {
            int nloc = bn * 128 + wn * 32 + tn * 16 + l15;
            float bv = bias[nloc];
            #pragma unroll
            for (int r = 0; r < 8; r++) {
                int mloc = bm * 128 + wm * 64 + tm * 16 + r + hi * 8;
                if (mloc < M) {
                    float v = acc[tm * 2 + tn][r] + bv;
                    size_t off = (size_t)mloc * N + nloc;
                    if (MODE == 0) outF[off] = v;
                    else if (MODE == 1) outF[off] = v + resid[off];
                    else if (MODE == 2) {
                        float g = v / (1.f + __expf(-1.702f * v));
                        outB[off] = f2bf(g);
                    } else outB[off] = f2bf(v);
                }
            }
        }
}

// ---------------- temporal attention (seq=8): one wave per (n,l,head) ----------------
__global__ __launch_bounds__(256) void attn_temporal(const uint16_t* __restrict__ qkv,
                                                     const float* __restrict__ rpb,
                                                     uint16_t* __restrict__ ato) {
    int wid = threadIdx.x >> 5, lane = threadIdx.x & 31;
    int pair = blockIdx.x * 8 + wid;           // 25088 pairs total
    int h = pair & 15;
    int nl = pair >> 4;
    int n = nl & 7, l = nl >> 3;
    int d0 = lane << 1;

    float q[8][2], k[8][2], vv[8][2];
    #pragma unroll
    for (int t = 0; t < 8; t++) {
        size_t base = (size_t)((l * TT + t) * NB + n) * 3072 + h * HD + d0;
        q[t][0]  = bf2f(qkv[base])        * 0.125f;
        q[t][1]  = bf2f(qkv[base + 1])    * 0.125f;
        k[t][0]  = bf2f(qkv[base + 1024]);
        k[t][1]  = bf2f(qkv[base + 1025]);
        vv[t][0] = bf2f(qkv[base + 2048]);
        vv[t][1] = bf2f(qkv[base + 2049]);
    }
    float s[8][8];
    #pragma unroll
    for (int tq = 0; tq < 8; tq++)
        #pragma unroll
        for (int tk = 0; tk < 8; tk++)
            s[tq][tk] = q[tq][0] * k[tk][0] + q[tq][1] * k[tk][1];
    #pragma unroll
    for (int m = 16; m >= 1; m >>= 1)
        #pragma unroll
        for (int tq = 0; tq < 8; tq++)
            #pragma unroll
            for (int tk = 0; tk < 8; tk++)
                s[tq][tk] += __shfl_xor(s[tq][tk], m, 32);

    #pragma unroll
    for (int tq = 0; tq < 8; tq++) {
        float mx = -1e30f;
        #pragma unroll
        for (int tk = 0; tk < 8; tk++) {
            s[tq][tk] += rpb[(tq - tk + 7) * NHEAD + h];
            mx = fmaxf(mx, s[tq][tk]);
        }
        float sum = 0.f;
        #pragma unroll
        for (int tk = 0; tk < 8; tk++) { s[tq][tk] = __expf(s[tq][tk] - mx); sum += s[tq][tk]; }
        float inv = 1.f / sum;
        float o0 = 0.f, o1 = 0.f;
        #pragma unroll
        for (int tk = 0; tk < 8; tk++) { o0 += s[tq][tk] * vv[tk][0]; o1 += s[tq][tk] * vv[tk][1]; }
        size_t orow = (size_t)((l * TT + tq) * NB + n) * DM + h * HD + d0;
        ato[orow]     = f2bf(o0 * inv);
        ato[orow + 1] = f2bf(o1 * inv);
    }
}

// ---------------- spatial attention (seq=197): WMMA flash per (b,h) ----------------
#define SP   224
#define KSTR 36     // uint stride (144B, 16B-aligned for async, conflict-free mod 64)
#define SST  224    // f32 stride, score block
#define PSTR 113    // uint stride, P block
#define VSTR 226    // bf16 stride, V^T

__global__ __launch_bounds__(256) void attn_spatial(const uint16_t* __restrict__ qkv,
                                                    const float* __restrict__ rpb,
                                                    uint16_t* __restrict__ ato) {
    __shared__ uint32_t Kt[SP * KSTR];        // 32256 B
    __shared__ uint32_t VtTu[HD * (VSTR/2)];  // 28928 B (bf16 V^T [64][226])
    __shared__ float    Sb[16 * SST];         // 14336 B
    __shared__ uint32_t Pb[16 * PSTR];        //  7232 B
    __shared__ uint32_t Qc[16 * KSTR];        //  2304 B

    int tid = threadIdx.x, wid = tid >> 5, lane = tid & 31;
    int l15 = lane & 15, hi = lane >> 4;
    int h = blockIdx.x & 15, b = blockIdx.x >> 4;
    const uint32_t* qkvu = (const uint32_t*)qkv;  // 1536 uints per token row
    uint16_t* VtT = (uint16_t*)VtTu;

    // stage K via async global->LDS (16B chunks), zero the padded rows
    for (int ch = tid; ch < 197 * 8; ch += 256) {
        int s = ch >> 3, jq = ch & 7;
        uint32_t lk = (uint32_t)(uintptr_t)&Kt[s * KSTR + jq * 4];
        async_ld16(lk, qkvu + (size_t)(b * 197 + s) * 1536 + 512 + h * 32 + jq * 4);
    }
    for (int u = tid; u < (SP - 197) * 32; u += 256) {
        int s = 197 + (u >> 5), j = u & 31;
        Kt[s * KSTR + j] = 0u;
    }
    // stage V transposed (elementwise scatter)
    for (int e = tid; e < SP * HD; e += 256) {
        int s = e >> 6, d = e & 63;
        uint16_t val = 0;
        if (s < 197) val = qkv[(size_t)(b * 197 + s) * 3072 + 2048 + h * HD + d];
        VtT[d * VSTR + s] = val;
    }
    wait_async0();

    for (int qc = 0; qc < 13; qc++) {
        int q0 = qc * 16;
        __syncthreads();
        for (int u = tid; u < 16 * 32; u += 256) {
            int m = u >> 5, j = u & 31;
            int sq = q0 + m;
            uint32_t val = 0u;
            if (sq < 197) val = qkvu[(size_t)(b * 197 + sq) * 1536 + h * 32 + j];
            Qc[m * KSTR + j] = val;
        }
        __syncthreads();

        // scores: S[16][224] = (Qc * 0.125) * Kt^T  via WMMA NT
        for (int nt = wid; nt < 14; nt += 8) {
            AccU c;
            #pragma unroll
            for (int r = 0; r < 8; r++) c.f[r] = 0.f;
            #pragma unroll
            for (int k0 = 0; k0 < 2; k0++) {
                FragU a, bb;
                #pragma unroll
                for (int v = 0; v < 8; v++) a.u[v] = Qc[l15 * KSTR + k0 * 16 + kpA(v, hi)];
                #pragma unroll
                for (int v = 0; v < 8; v++) bb.u[v] = Kt[(nt * 16 + l15) * KSTR + k0 * 16 + kpB(v, hi)];
                c.v = __builtin_amdgcn_wmma_f32_16x16x32_bf16(
                          false, a.v, false, bb.v, (short)0, c.v, false, false);
            }
            #pragma unroll
            for (int r = 0; r < 8; r++)
                Sb[(r + 8 * hi) * SST + nt * 16 + l15] = c.f[r] * 0.125f;
        }
        __syncthreads();

        // softmax + relative position bias -> P (bf16)
        {
            int m = tid >> 4, c0 = tid & 15;
            int sq = q0 + m;
            float vals[14];
            float mx = -1e30f;
            #pragma unroll
            for (int jj = 0; jj < 14; jj++) {
                int cc = c0 * 14 + jj;
                float sv = Sb[m * SST + cc];
                if (cc >= 197) sv = -1e30f;
                else if (sq >= 1 && sq < 197 && cc >= 1) {
                    int lq = sq - 1, lk = cc - 1;
                    int qi = lq / SS, qj = lq % SS, ki = lk / SS, kj = lk % SS;
                    int idx = (qi - ki + SS - 1) * (2 * SS - 1) + (qj - kj + SS - 1);
                    sv += rpb[idx * NHEAD + h];
                }
                vals[jj] = sv;
                mx = fmaxf(mx, sv);
            }
            #pragma unroll
            for (int msk = 8; msk >= 1; msk >>= 1) mx = fmaxf(mx, __shfl_xor(mx, msk, 32));
            float sum = 0.f;
            #pragma unroll
            for (int jj = 0; jj < 14; jj++) { vals[jj] = __expf(vals[jj] - mx); sum += vals[jj]; }
            #pragma unroll
            for (int msk = 8; msk >= 1; msk >>= 1) sum += __shfl_xor(sum, msk, 32);
            float inv = 1.f / sum;
            #pragma unroll
            for (int pj = 0; pj < 7; pj++) {
                uint32_t p0 = f2bf(vals[2 * pj] * inv);
                uint32_t p1 = f2bf(vals[2 * pj + 1] * inv);
                Pb[m * PSTR + c0 * 7 + pj] = p0 | (p1 << 16);
            }
        }
        __syncthreads();

        // AV: O[16][64] = P[16][224] * V[224][64]  via WMMA NN (V^T staged)
        if (wid < 4) {
            int tn = wid;
            AccU c;
            #pragma unroll
            for (int r = 0; r < 8; r++) c.f[r] = 0.f;
            for (int k0 = 0; k0 < SP; k0 += 32) {
                FragU a, bb;
                #pragma unroll
                for (int v = 0; v < 8; v++) a.u[v] = Pb[l15 * PSTR + (k0 >> 1) + kpA(v, hi)];
                int col = tn * 16 + l15;
                #pragma unroll
                for (int v = 0; v < 8; v++)
                    bb.u[v] = VtTu[col * (VSTR / 2) + (k0 >> 1) + kpB(v, hi)];
                c.v = __builtin_amdgcn_wmma_f32_16x16x32_bf16(
                          false, a.v, false, bb.v, (short)0, c.v, false, false);
            }
            #pragma unroll
            for (int r = 0; r < 8; r++) {
                int sq = q0 + r + 8 * hi;
                if (sq < 197)
                    ato[(size_t)(b * 197 + sq) * DM + h * HD + tn * 16 + l15] = f2bf(c.f[r]);
            }
        }
    }
}

// ---------------- final combine: XO = Xc + concat(mean_t cls, spa) ----------------
__global__ __launch_bounds__(256) void combine_out(const float* __restrict__ Xc,
                                                   const float* __restrict__ RS,
                                                   float* __restrict__ XO) {
    int ro = blockIdx.x;               // 0..12551
    int tid = threadIdx.x;
    if (ro < NB) {
        int n = ro;
        #pragma unroll
        for (int i = 0; i < 4; i++) {
            int c = tid + i * 256;
            float acc = 0.f;
            #pragma unroll
            for (int t = 0; t < TT; t++)
                acc += RS[(size_t)((t * NB + n) * 197) * DM + c];
            XO[(size_t)ro * DM + c] = Xc[(size_t)ro * DM + c] + 0.125f * acc;
        }
    } else {
        int r = ro - NB;
        int n = r & 7, lt = r >> 3;
        int t = lt & 7, l = lt >> 3;
        size_t src = (size_t)((t * NB + n) * 197 + 1 + l) * DM;
        #pragma unroll
        for (int i = 0; i < 4; i++) {
            int c = tid + i * 256;
            XO[(size_t)ro * DM + c] = Xc[(size_t)ro * DM + c] + RS[src + c];
        }
    }
}

// ---------------- host orchestration ----------------
extern "C" void kernel_launch(void* const* d_in, const int* in_sizes, int n_in,
                              void* d_out, int out_size, void* d_ws, size_t ws_size,
                              hipStream_t stream) {
    const float* x      = (const float*)d_in[0];
    const float* pos_w  = (const float*)d_in[1];
    const float* pos_b  = (const float*)d_in[2];
    const float* ln_t_g = (const float*)d_in[3];
    const float* ln_t_b = (const float*)d_in[4];
    const float* rpb_t  = (const float*)d_in[5];
    const float* wqkv_t = (const float*)d_in[6];
    const float* bqkv_t = (const float*)d_in[7];
    const float* wo_t   = (const float*)d_in[8];
    const float* bo_t   = (const float*)d_in[9];
    const float* ln1_g  = (const float*)d_in[10];
    const float* ln1_b  = (const float*)d_in[11];
    const float* rpb_s  = (const float*)d_in[12];
    const float* wqkv_s = (const float*)d_in[13];
    const float* bqkv_s = (const float*)d_in[14];
    const float* wo_s   = (const float*)d_in[15];
    const float* bo_s   = (const float*)d_in[16];
    const float* ln2_g  = (const float*)d_in[17];
    const float* ln2_b  = (const float*)d_in[18];
    const float* fc_w   = (const float*)d_in[19];
    const float* fc_b   = (const float*)d_in[20];
    const float* proj_w = (const float*)d_in[21];
    const float* proj_b = (const float*)d_in[22];

    char* ws = (char*)d_ws;
    size_t off = 0;
    auto alloc = [&](size_t bytes) -> char* {
        off = (off + 255) & ~(size_t)255;
        char* p = ws + off;
        off += bytes;
        return p;
    };

    float*    Xc      = (float*)alloc((size_t)MALL * DM * 4);
    uint16_t* wqkvt_b = (uint16_t*)alloc((size_t)3072 * DM * 2);
    uint16_t* wot_b   = (uint16_t*)alloc((size_t)DM * DM * 2);
    uint16_t* wqkvs_b = (uint16_t*)alloc((size_t)3072 * DM * 2);
    uint16_t* wos_b   = (uint16_t*)alloc((size_t)DM * DM * 2);
    uint16_t* fc_b16  = (uint16_t*)alloc((size_t)4096 * DM * 2);
    uint16_t* pj_b16  = (uint16_t*)alloc((size_t)DM * 4096 * 2);
    uint16_t* actA    = (uint16_t*)alloc((size_t)MQ_S * DM * 2);   // XT / XS / XN2
    uint16_t* big     = (uint16_t*)alloc((size_t)MALL * 4096 * 2); // QKV / H(gelu)
    uint16_t* ato     = (uint16_t*)alloc((size_t)MQ_S * DM * 2);   // attn outputs
    float*    f32buf  = (float*)alloc((size_t)MQ_S * DM * 4);      // XTF then RS
    float*    XO      = (float*)d_out;
    (void)ws_size; (void)in_sizes; (void)n_in; (void)out_size;

    // weights -> bf16
    cast_f32_bf16<<<(3145728 + 255) / 256, 256, 0, stream>>>(wqkv_t, wqkvt_b, 3145728);
    cast_f32_bf16<<<(1048576 + 255) / 256, 256, 0, stream>>>(wo_t,   wot_b,   1048576);
    cast_f32_bf16<<<(3145728 + 255) / 256, 256, 0, stream>>>(wqkv_s, wqkvs_b, 3145728);
    cast_f32_bf16<<<(1048576 + 255) / 256, 256, 0, stream>>>(wo_s,   wos_b,   1048576);
    cast_f32_bf16<<<(4194304 + 255) / 256, 256, 0, stream>>>(fc_w,   fc_b16,  4194304);
    cast_f32_bf16<<<(4194304 + 255) / 256, 256, 0, stream>>>(proj_w, pj_b16,  4194304);

    // positional conv embedding (+ cls copy)
    copy_f32<<<32, 256, 0, stream>>>(x, Xc, NB * DM);
    conv_pos<<<(int)(((size_t)LT * NB * DM) / 256), 256, 0, stream>>>(x, pos_w, pos_b, Xc);

    // ---- temporal attention ----
    ln_rows<<<MQ_T, 256, 0, stream>>>(Xc + (size_t)NB * DM, ln_t_g, ln_t_b, actA, MQ_T);
    gemm_nt<3><<<dim3(24, 98), 256, 0, stream>>>(actA, wqkvt_b, bqkv_t, nullptr,
                                                 nullptr, big, MQ_T, 3072, DM);
    attn_temporal<<<3136, 256, 0, stream>>>(big, rpb_t, ato);
    gemm_nt<1><<<dim3(8, 98), 256, 0, stream>>>(ato, wot_b, bo_t, Xc + (size_t)NB * DM,
                                                f32buf, nullptr, MQ_T, DM, DM);  // XTF

    // ---- spatial attention ----
    pack_xs_ln<<<MQ_S, 256, 0, stream>>>(Xc, f32buf, ln1_g, ln1_b, actA);
    gemm_nt<3><<<dim3(24, 99), 256, 0, stream>>>(actA, wqkvs_b, bqkv_s, nullptr,
                                                 nullptr, big, MQ_S, 3072, DM);
    attn_spatial<<<1024, 256, 0, stream>>>(big, rpb_s, ato);
    gemm_nt<0><<<dim3(8, 99), 256, 0, stream>>>(ato, wos_b, bo_s, nullptr,
                                                f32buf, nullptr, MQ_S, DM, DM);  // RS
    combine_out<<<MALL, 256, 0, stream>>>(Xc, f32buf, XO);

    // ---- MLP ----
    ln_rows<<<MALL, 256, 0, stream>>>(XO, ln2_g, ln2_b, actA, MALL);
    gemm_nt<2><<<dim3(32, 99), 256, 0, stream>>>(actA, fc_b16, fc_b, nullptr,
                                                 nullptr, big, MALL, 4096, DM);
    gemm_nt<1><<<dim3(8, 99), 256, 0, stream>>>(big, pj_b16, proj_b, XO,
                                                XO, nullptr, MALL, DM, 4096);
}
